// Model16_9620726743229
// MI455X (gfx1250) — matrix operand order repaired and checked by reading, not compiled, not run on hardware
//
#include <hip/hip_runtime.h>

// ---------------- types ----------------
typedef __bf16  bf16x16 __attribute__((ext_vector_type(16)));
typedef float   f32x8   __attribute__((ext_vector_type(8)));
typedef unsigned int u32x4 __attribute__((ext_vector_type(4)));
typedef int     v4i_gnu __attribute__((vector_size(16)));   // matches async-LDS builtin param

union Frag {
    bf16x16 v;
    u32x4   q[2];
};

__device__ __forceinline__ unsigned short f2bf(float f) {
    unsigned int u = __float_as_uint(f);
    unsigned int r = u + 0x7FFFu + ((u >> 16) & 1u);
    return (unsigned short)(r >> 16);
}
__device__ __forceinline__ float bf2f(unsigned short h) {
    return __uint_as_float(((unsigned int)h) << 16);
}

__device__ __forceinline__ f32x8 wmma_bf16(const Frag& a, const Frag& b, f32x8 c) {
    // (neg_a, A, neg_b, B, c_mod, C, reuse_a, reuse_b)
    return __builtin_amdgcn_wmma_f32_16x16x32_bf16(false, a.v, false, b.v,
                                                   (short)0, c, false, false);
}

// ---- CDNA5 async global->LDS copy (ASYNCcnt path), guarded ----
#if defined(__has_builtin)
#if __has_builtin(__builtin_amdgcn_global_load_async_to_lds_b128)
#define HAVE_ASYNC_LDS 1
#endif
#endif
#ifndef HAVE_ASYNC_LDS
#define HAVE_ASYNC_LDS 0
#endif

#if HAVE_ASYNC_LDS
__device__ __forceinline__ void async_copy16(const unsigned short* g, unsigned short* l) {
    __builtin_amdgcn_global_load_async_to_lds_b128(
        (__attribute__((address_space(1))) v4i_gnu*)g,
        (__attribute__((address_space(3))) v4i_gnu*)l, 0, 0);
}
__device__ __forceinline__ void wait_async() {
#if __has_builtin(__builtin_amdgcn_s_wait_asynccnt)
    __builtin_amdgcn_s_wait_asynccnt(0);
#else
    asm volatile("s_wait_asynccnt 0x0" ::: "memory");
#endif
}
#endif

// ---------------- constants ----------------
#define BATCH   2048
#define NODES   20
#define U3      64
#define XDIM    69          // 64 emb + 5 raw
#define XSTRIDE 72          // padded x_cat row stride (bf16 elems)
#define NM      16

// Edge A-row layout (column-permuted so gathers are 16B-aligned chunks):
//   hasDst: src feats @ [0,72), dst feats @ [72,144), armies @ 144, pad to 160
//   !hasDst: src feats @ [0,72), armies @ 72, pad to 96
// Weight transpose applies the identical permutation.

// ---------------- kernel: pack x105 -> bf16 [B,128] ----------------
__global__ __launch_bounds__(256) void k_pack_x105(
    const float* __restrict__ gf, const float* __restrict__ income,
    const float* __restrict__ tot, unsigned short* __restrict__ X) {
    int i = blockIdx.x * 256 + threadIdx.x;           // over 2048*128
    int b = i >> 7, c = i & 127;
    float v = 0.f;
    if (c < 100)       v = gf[b * 100 + c];
    else if (c < 102)  v = income[b * 2 + (c - 100)];
    else if (c < 105)  v = tot[b * 3 + (c - 102)];
    X[i] = f2bf(v);
}

// ---------------- kernel: W [K,N] f32 -> Wt [N,Kpad] bf16 (transposed, pad) ----
__global__ __launch_bounds__(256) void k_w_transpose(
    const float* __restrict__ W, unsigned short* __restrict__ Wt,
    int K, int N, int Kpad) {
    long long i = (long long)blockIdx.x * 256 + threadIdx.x;
    long long total = (long long)N * Kpad;
    if (i >= total) return;
    int n = (int)(i / Kpad), k = (int)(i % Kpad);
    Wt[i] = (k < K) ? f2bf(W[(long long)k * N + n]) : (unsigned short)0;
}

// ---- edge layer-1 weight transpose with the column permutation above ----
__global__ __launch_bounds__(256) void k_w_transpose_edge(
    const float* __restrict__ W, unsigned short* __restrict__ Wt,
    int N, int Kpad, int hasDst) {
    long long i = (long long)blockIdx.x * 256 + threadIdx.x;
    if (i >= (long long)N * Kpad) return;
    int n = (int)(i / Kpad), nk = (int)(i % Kpad);
    int ok = -1;
    if (nk < XDIM)                                  ok = 1 + nk;               // src feats
    else if (hasDst && nk >= 72 && nk < 72 + XDIM)  ok = 1 + XDIM + (nk - 72); // dst feats
    else if (nk == (hasDst ? 144 : 72))             ok = 0;                    // armies
    Wt[i] = (ok >= 0) ? f2bf(W[(long long)ok * N + n]) : (unsigned short)0;
}

// ---------------- kernel: generic bf16 WMMA GEMM, 16x64 tile per wave ----------
// out[M,N] = act(A[M,Kpad] @ Wt[N,Kpad]^T + bias); N % 64 == 0, M % 16 == 0.
__global__ __launch_bounds__(256) void k_gemm_bf16(
    const unsigned short* __restrict__ A, const unsigned short* __restrict__ Wt,
    const float* __restrict__ bias, unsigned short* __restrict__ out,
    int M, int N, int Kpad, int relu) {
    int wid  = threadIdx.x >> 5, lane = threadIdx.x & 31;
    int tile = blockIdx.x * 8 + wid;
    int tilesN = N >> 6;
    int tm = tile / tilesN, tn = tile - tm * tilesN;
    if (tm * 16 >= M) return;
    int m0 = tm * 16, n0 = tn * 64;
    int l15 = lane & 15, half = lane >> 4;

    const unsigned short* arow = A  + (size_t)(m0 + l15) * Kpad + half * 8;
    const unsigned short* brow = Wt + (size_t)(n0 + l15) * Kpad + half * 8;
    size_t bs = (size_t)16 * Kpad;

    f32x8 c0 = {0.f, 0.f, 0.f, 0.f, 0.f, 0.f, 0.f, 0.f};
    f32x8 c1 = c0, c2 = c0, c3 = c0;
    for (int k0 = 0; k0 < Kpad; k0 += 32) {
        __builtin_prefetch(arow + k0 + 256, 0, 0);
        __builtin_prefetch(brow + k0 + 256, 0, 0);
        Frag a, b;
        a.q[0] = *(const u32x4*)(arow + k0);
        a.q[1] = *(const u32x4*)(arow + k0 + 16);
        b.q[0] = *(const u32x4*)(brow + k0);
        b.q[1] = *(const u32x4*)(brow + k0 + 16);
        c0 = wmma_bf16(a, b, c0);
        b.q[0] = *(const u32x4*)(brow + bs + k0);
        b.q[1] = *(const u32x4*)(brow + bs + k0 + 16);
        c1 = wmma_bf16(a, b, c1);
        b.q[0] = *(const u32x4*)(brow + 2 * bs + k0);
        b.q[1] = *(const u32x4*)(brow + 2 * bs + k0 + 16);
        c2 = wmma_bf16(a, b, c2);
        b.q[0] = *(const u32x4*)(brow + 3 * bs + k0);
        b.q[1] = *(const u32x4*)(brow + 3 * bs + k0 + 16);
        c3 = wmma_bf16(a, b, c3);
    }
    int rb = half * 8;
    auto store_tile = [&](f32x8 c, int j) {
        float bv = bias[n0 + j * 16 + l15];
#pragma unroll
        for (int i = 0; i < 8; ++i) {
            float v = c[i] + bv;
            if (relu) v = fmaxf(v, 0.f);
            out[(size_t)(m0 + rb + i) * N + n0 + j * 16 + l15] = f2bf(v);
        }
    };
    store_tile(c0, 0); store_tile(c1, 1); store_tile(c2, 2); store_tile(c3, 3);
}

// ---------------- kernel: build x_cat bf16 [40960, 72] ----------------
__global__ __launch_bounds__(256) void k_build_xcat(
    const unsigned short* __restrict__ H3, const float* __restrict__ gf,
    unsigned short* __restrict__ xcat) {
    long long i = (long long)blockIdx.x * 256 + threadIdx.x;   // over 40960*72
    if (i >= (long long)BATCH * NODES * XSTRIDE) return;
    int g = (int)(i / XSTRIDE), c = (int)(i % XSTRIDE);
    unsigned short v = 0;
    if (c < U3)        v = H3[(size_t)(g / NODES) * 1280 + (g % NODES) * U3 + c];
    else if (c < XDIM) v = f2bf(gf[(size_t)g * 5 + (c - U3)]);
    xcat[i] = v;
}

// ---------------- kernel: zero pooled ----------------
__global__ __launch_bounds__(256) void k_zero(float* __restrict__ p, int n) {
    int i = blockIdx.x * 256 + threadIdx.x;
    if (i < n) p[i] = 0.f;
}

// ---------------- kernel: fused 3-layer edge MLP + segment add --------------
// 2 waves/block, each wave owns a 16-edge tile; edge pipeline lives in LDS.
template <int K1PAD, int HAS_DST>
__global__ __launch_bounds__(64) void k_edge(
    const float* __restrict__ armies, const int* __restrict__ src,
    const int* __restrict__ dst, const int* __restrict__ btch,
    const unsigned short* __restrict__ xcat,
    const unsigned short* __restrict__ W1t, const float* __restrict__ b1v,
    const unsigned short* __restrict__ W2t, const float* __restrict__ b2v,
    const unsigned short* __restrict__ Wot, const float* __restrict__ bov,
    float* __restrict__ pooled) {
    __shared__ __align__(16) unsigned short A1[2][16 * K1PAD];
    __shared__ __align__(16) unsigned short H1s[2][16 * 256];
    __shared__ __align__(16) unsigned short H2s[2][16 * 256];
    __shared__ int bidxs[2][16];

    constexpr int ACOL = HAS_DST ? 144 : 72;   // armies column
    int wid = threadIdx.x >> 5, lane = threadIdx.x & 31;
    int e0  = (blockIdx.x * 2 + wid) * 16;     // E % 32 == 0, always full
    int l15 = lane & 15, half = lane >> 4;

    if (lane < 16) bidxs[wid][lane] = btch[e0 + lane];

#if HAVE_ASYNC_LDS
    // Gather x_cat rows as 9x16B async chunks per row (72 elems incl. zero pad).
    for (int i = lane; i < 16 * 9; i += 32) {
        int r = i / 9, ch = i - r * 9;
        async_copy16(xcat + (size_t)src[e0 + r] * XSTRIDE + ch * 8,
                     &A1[wid][r * K1PAD + ch * 8]);
    }
    if (HAS_DST) {
        for (int i = lane; i < 16 * 9; i += 32) {
            int r = i / 9, ch = i - r * 9;
            async_copy16(xcat + (size_t)dst[e0 + r] * XSTRIDE + ch * 8,
                         &A1[wid][r * K1PAD + 72 + ch * 8]);
        }
    }
#else
    for (int r = 0; r < 16; ++r) {
        int srow = src[e0 + r] * XSTRIDE;
        int drow = HAS_DST ? dst[e0 + r] * XSTRIDE : 0;
        for (int c = lane; c < ACOL; c += 32) {
            unsigned short v = (c < 72) ? xcat[srow + c] : xcat[drow + (c - 72)];
            A1[wid][r * K1PAD + c] = v;
        }
    }
#endif
    // armies + tail padding (disjoint from async destination range)
    for (int r = 0; r < 16; ++r) {
        unsigned short arm = f2bf(armies[e0 + r]);
        for (int c = ACOL + lane; c < K1PAD; c += 32)
            A1[wid][r * K1PAD + c] = (c == ACOL) ? arm : (unsigned short)0;
    }
#if HAVE_ASYNC_LDS
    wait_async();
#endif
    __syncthreads();

    // ---- layer 1: [16,K1PAD] @ W1t -> relu -> H1s [16,256] ----
    {
        constexpr int NK = K1PAD / 32;
        const unsigned short* abase = &A1[wid][0] + l15 * K1PAD + half * 8;
        Frag afr[NK];
#pragma unroll
        for (int kk = 0; kk < NK; ++kk) {
            afr[kk].q[0] = *(const u32x4*)(abase + kk * 32);
            afr[kk].q[1] = *(const u32x4*)(abase + kk * 32 + 16);
        }
        for (int nt = 0; nt < 16; ++nt) {
            int n0 = nt * 16;
            const unsigned short* brow = W1t + (size_t)(n0 + l15) * K1PAD + half * 8;
            f32x8 c = {0.f, 0.f, 0.f, 0.f, 0.f, 0.f, 0.f, 0.f};
#pragma unroll
            for (int kk = 0; kk < NK; ++kk) {
                Frag b;
                b.q[0] = *(const u32x4*)(brow + kk * 32);
                b.q[1] = *(const u32x4*)(brow + kk * 32 + 16);
                c = wmma_bf16(afr[kk], b, c);
            }
            float bv = b1v[n0 + l15];
            int rb = half * 8;
#pragma unroll
            for (int i = 0; i < 8; ++i)
                H1s[wid][(rb + i) * 256 + n0 + l15] = f2bf(fmaxf(c[i] + bv, 0.f));
        }
    }
    __syncthreads();

    // ---- layer 2: H1s @ W2t -> relu -> H2s ----
    {
        const unsigned short* abase = &H1s[wid][0] + l15 * 256 + half * 8;
        Frag afr[8];
#pragma unroll
        for (int kk = 0; kk < 8; ++kk) {
            afr[kk].q[0] = *(const u32x4*)(abase + kk * 32);
            afr[kk].q[1] = *(const u32x4*)(abase + kk * 32 + 16);
        }
        for (int nt = 0; nt < 16; ++nt) {
            int n0 = nt * 16;
            const unsigned short* brow = W2t + (size_t)(n0 + l15) * 256 + half * 8;
            f32x8 c = {0.f, 0.f, 0.f, 0.f, 0.f, 0.f, 0.f, 0.f};
#pragma unroll
            for (int kk = 0; kk < 8; ++kk) {
                Frag b;
                b.q[0] = *(const u32x4*)(brow + kk * 32);
                b.q[1] = *(const u32x4*)(brow + kk * 32 + 16);
                c = wmma_bf16(afr[kk], b, c);
            }
            float bv = b2v[n0 + l15];
            int rb = half * 8;
#pragma unroll
            for (int i = 0; i < 8; ++i)
                H2s[wid][(rb + i) * 256 + n0 + l15] = f2bf(fmaxf(c[i] + bv, 0.f));
        }
    }
    __syncthreads();

    // ---- order layer: H2s @ Wot + bo -> atomic segment add into pooled ----
    {
        const unsigned short* abase = &H2s[wid][0] + l15 * 256 + half * 8;
        Frag afr[8];
#pragma unroll
        for (int kk = 0; kk < 8; ++kk) {
            afr[kk].q[0] = *(const u32x4*)(abase + kk * 32);
            afr[kk].q[1] = *(const u32x4*)(abase + kk * 32 + 16);
        }
        for (int nt = 0; nt < 16; ++nt) {
            int n0 = nt * 16;
            const unsigned short* brow = Wot + (size_t)(n0 + l15) * 256 + half * 8;
            f32x8 c = {0.f, 0.f, 0.f, 0.f, 0.f, 0.f, 0.f, 0.f};
#pragma unroll
            for (int kk = 0; kk < 8; ++kk) {
                Frag b;
                b.q[0] = *(const u32x4*)(brow + kk * 32);
                b.q[1] = *(const u32x4*)(brow + kk * 32 + 16);
                c = wmma_bf16(afr[kk], b, c);
            }
            float bv = bov[n0 + l15];
            int rb = half * 8;
#pragma unroll
            for (int i = 0; i < 8; ++i) {
                int bidx = bidxs[wid][rb + i];
                atomicAdd(&pooled[(size_t)bidx * 256 + n0 + l15], c[i] + bv);
            }
        }
    }
}

// ---------------- kernel: pooled -> relu -> @Wf+bf -> log_softmax(16) ----------
__global__ __launch_bounds__(256) void k_pool_head(
    const float* __restrict__ pooled, const float* __restrict__ Wf,
    const float* __restrict__ bfv, float* __restrict__ pi) {
    int b = blockIdx.x, t = threadIdx.x;
    __shared__ float red[256];
    __shared__ float tm[NM];
    for (int m = 0; m < NM; ++m) {
        float v = pooled[((size_t)(b * NM + m)) * 256 + t];
        red[t] = fmaxf(v, 0.f) * Wf[t];
        __syncthreads();
        for (int s = 128; s > 0; s >>= 1) {
            if (t < s) red[t] += red[t + s];
            __syncthreads();
        }
        if (t == 0) tm[m] = red[0] + bfv[0];
        __syncthreads();
    }
    if (t == 0) {
        float mx = tm[0];
        for (int m = 1; m < NM; ++m) mx = fmaxf(mx, tm[m]);
        float se = 0.f;
        for (int m = 0; m < NM; ++m) se += expf(tm[m] - mx);
        float ls = logf(se);
        for (int m = 0; m < NM; ++m) pi[b * NM + m] = tm[m] - mx - ls;
    }
}

// ---------------- kernel: value head val = tanh(H3 @ W4 + b4) ----------------
__global__ __launch_bounds__(256) void k_val_head(
    const unsigned short* __restrict__ H3, const float* __restrict__ W4,
    const float* __restrict__ b4, float* __restrict__ out) {
    int b = blockIdx.x, t = threadIdx.x;
    __shared__ float red[256];
    float s = 0.f;
    for (int k = t; k < 1280; k += 256) s += bf2f(H3[(size_t)b * 1280 + k]) * W4[k];
    red[t] = s;
    __syncthreads();
    for (int st = 128; st > 0; st >>= 1) {
        if (t < st) red[t] += red[t + st];
        __syncthreads();
    }
    if (t == 0) out[b] = tanhf(red[0] + b4[0]);
}

// ---------------- host launcher ----------------
extern "C" void kernel_launch(void* const* d_in, const int* in_sizes, int n_in,
                              void* d_out, int out_size, void* d_ws, size_t ws_size,
                              hipStream_t stream) {
    const float* gf     = (const float*)d_in[0];
    const float* income = (const float*)d_in[1];
    const float* tot    = (const float*)d_in[2];
    const float* aar    = (const float*)d_in[3];
    const float* tar    = (const float*)d_in[4];
    const float* dar    = (const float*)d_in[5];
    const float* W1 = (const float*)d_in[6];  const float* b1 = (const float*)d_in[7];
    const float* W2 = (const float*)d_in[8];  const float* b2 = (const float*)d_in[9];
    const float* W3 = (const float*)d_in[10]; const float* b3 = (const float*)d_in[11];
    const float* W4 = (const float*)d_in[12]; const float* b4 = (const float*)d_in[13];
    const float* Wa = (const float*)d_in[14]; const float* ba = (const float*)d_in[15];
    const float* Wa2= (const float*)d_in[16]; const float* ba2= (const float*)d_in[17];
    const float* Wtr= (const float*)d_in[18]; const float* bt = (const float*)d_in[19];
    const float* Wt2= (const float*)d_in[20]; const float* bt2= (const float*)d_in[21];
    const float* Wd = (const float*)d_in[22]; const float* bd = (const float*)d_in[23];
    const float* Wd2= (const float*)d_in[24]; const float* bd2= (const float*)d_in[25];
    const float* Wo = (const float*)d_in[26]; const float* bo = (const float*)d_in[27];
    const float* Wf = (const float*)d_in[28]; const float* bfv= (const float*)d_in[29];
    const int* asrcs = (const int*)d_in[30];
    const int* adsts = (const int*)d_in[31];
    const int* tsrcs = (const int*)d_in[32];
    const int* tdsts = (const int*)d_in[33];
    const int* dtgts = (const int*)d_in[34];
    const int* abtch = (const int*)d_in[35];
    const int* tbtch = (const int*)d_in[36];
    const int* dbtch = (const int*)d_in[37];

    const int E = in_sizes[3];               // 98304

    char* wsp = (char*)d_ws;
    size_t off = 0;
    auto alloc = [&](size_t bytes) {
        void* p = wsp + off;
        off = (off + bytes + 255) & ~(size_t)255;
        return p;
    };
    unsigned short* X105 = (unsigned short*)alloc((size_t)2048 * 128 * 2);
    unsigned short* W1t  = (unsigned short*)alloc((size_t)2048 * 128 * 2);
    unsigned short* H1   = (unsigned short*)alloc((size_t)2048 * 2048 * 2);
    unsigned short* W2t  = (unsigned short*)alloc((size_t)2048 * 2048 * 2);
    unsigned short* H2   = (unsigned short*)alloc((size_t)2048 * 2048 * 2);
    unsigned short* W3t  = (unsigned short*)alloc((size_t)1280 * 2048 * 2);
    unsigned short* H3   = (unsigned short*)alloc((size_t)2048 * 1280 * 2);
    unsigned short* XC   = (unsigned short*)alloc((size_t)2048 * NODES * XSTRIDE * 2);
    unsigned short* WaT  = (unsigned short*)alloc((size_t)256 * 160 * 2);
    unsigned short* WtT  = (unsigned short*)alloc((size_t)256 * 160 * 2);
    unsigned short* WdT  = (unsigned short*)alloc((size_t)256 * 96 * 2);
    unsigned short* Wa2T = (unsigned short*)alloc((size_t)256 * 256 * 2);
    unsigned short* Wt2T = (unsigned short*)alloc((size_t)256 * 256 * 2);
    unsigned short* Wd2T = (unsigned short*)alloc((size_t)256 * 256 * 2);
    unsigned short* WoT  = (unsigned short*)alloc((size_t)256 * 256 * 2);
    float* pooled        = (float*)alloc((size_t)2048 * NM * 256 * 4);

    // 1) pack x105 -> bf16 [2048,128]
    k_pack_x105<<<(2048 * 128) / 256, 256, 0, stream>>>(gf, income, tot, X105);

    // 2) transpose + cast all GEMM weights to [N,Kpad] bf16
    auto launch_tr = [&](const float* W, unsigned short* Wt_, int K, int N, int Kpad) {
        long long total = (long long)N * Kpad;
        k_w_transpose<<<(int)((total + 255) / 256), 256, 0, stream>>>(W, Wt_, K, N, Kpad);
    };
    launch_tr(W1, W1t, 105, 2048, 128);
    launch_tr(W2, W2t, 2048, 2048, 2048);
    launch_tr(W3, W3t, 2048, 1280, 2048);
    launch_tr(Wa2, Wa2T, 256, 256, 256);
    launch_tr(Wt2, Wt2T, 256, 256, 256);
    launch_tr(Wd2, Wd2T, 256, 256, 256);
    launch_tr(Wo,  WoT,  256, 256, 256);
    // edge layer-1 weights with the gather-friendly column permutation
    k_w_transpose_edge<<<(256 * 160) / 256, 256, 0, stream>>>(Wa, WaT, 256, 160, 1);
    k_w_transpose_edge<<<(256 * 160) / 256, 256, 0, stream>>>(Wtr, WtT, 256, 160, 1);
    k_w_transpose_edge<<<(256 * 96) / 256, 256, 0, stream>>>(Wd, WdT, 256, 96, 0);

    // 3) dense front-end (relu each layer); 16x64 tile per wave
    k_gemm_bf16<<<(128 * 32) / 8, 256, 0, stream>>>(X105, W1t, b1, H1, 2048, 2048, 128, 1);
    k_gemm_bf16<<<(128 * 32) / 8, 256, 0, stream>>>(H1, W2t, b2, H2, 2048, 2048, 2048, 1);
    k_gemm_bf16<<<(128 * 20) / 8, 256, 0, stream>>>(H2, W3t, b3, H3, 2048, 1280, 2048, 1);

    // 4) x_cat = [node emb (64) | raw features (5)] bf16, stride 72
    {
        long long total = (long long)2048 * NODES * XSTRIDE;
        k_build_xcat<<<(int)((total + 255) / 256), 256, 0, stream>>>(H3, gf, XC);
    }

    // 5) zero the pooled segment-sum buffer
    k_zero<<<(2048 * NM * 256) / 256, 256, 0, stream>>>(pooled, 2048 * NM * 256);

    // 6) fused edge MLPs (attack / transfer / deploy) with atomic segment-sum
    int eblocks = E / 32;
    k_edge<160, 1><<<eblocks, 64, 0, stream>>>(aar, asrcs, adsts, abtch, XC,
                                               WaT, ba, Wa2T, ba2, WoT, bo, pooled);
    k_edge<160, 1><<<eblocks, 64, 0, stream>>>(tar, tsrcs, tdsts, tbtch, XC,
                                               WtT, bt, Wt2T, bt2, WoT, bo, pooled);
    k_edge<96, 0><<<eblocks, 64, 0, stream>>>(dar, dtgts, (const int*)nullptr, dbtch, XC,
                                              WdT, bd, Wd2T, bd2, WoT, bo, pooled);

    // 7) heads: d_out = [val (2048 f32) | pi (2048*16 f32)]
    float* out_val = (float*)d_out;
    float* out_pi  = out_val + 2048;
    k_pool_head<<<2048, 256, 0, stream>>>(pooled, Wf, bfv, out_pi);
    k_val_head<<<2048, 256, 0, stream>>>(H3, W4, b4, out_val);
}
